// SeparableConv2d_Hyper_31868657336832
// MI455X (gfx1250) — matrix operand order repaired and checked by reading, not compiled, not run on hardware
//
#include <hip/hip_runtime.h>

typedef __attribute__((ext_vector_type(16))) _Float16 v16h;
typedef __attribute__((ext_vector_type(8)))  _Float16 half8;
typedef __attribute__((ext_vector_type(8)))  float    v8f;

#define CIN    256
#define COUT   256
#define HH     128
#define WW     128
#define TILE_N 64
#define KC     32
#define LDA    40   // halves per A row (32 data + 8 pad) -> 80B, 16B aligned, conflict-free
#define LDB    40   // halves per B row (per pixel)

__global__ __launch_bounds__(256)
void fused_dw_pw_wmma(const float* __restrict__ x,      // (B, Cin, H, W)
                      const float* __restrict__ delta,  // (B, Cout, Cin)
                      const float* __restrict__ dwk,    // (Cin, 9)
                      const float* __restrict__ pw,     // (Cout, Cin)
                      float* __restrict__ out)          // (B, Cout, H, W)
{
    __shared__ _Float16 Alds[COUT * LDA];    // 20 KB : weights chunk, [m][k]
    __shared__ _Float16 Blds[TILE_N * LDB];  //  5 KB : depthwise chunk, [n][k]

    const int b  = blockIdx.y;
    const int h  = blockIdx.x >> 1;
    const int w0 = (blockIdx.x & 1) * TILE_N;

    const int tid    = threadIdx.x;
    const int lane   = tid & 31;
    const int laneLo = lane & 15;
    const int laneHi = lane >> 4;
    const int wave   = tid >> 5;
    const int mWave  = (wave & 3) * 64;   // this wave's M range [mWave, mWave+64)
    const int nWave  = (wave >> 2) * 32;  // this wave's N range [nWave, nWave+32)

    v8f acc[4][2];
    #pragma unroll
    for (int i = 0; i < 4; ++i)
        #pragma unroll
        for (int j = 0; j < 2; ++j) {
            v8f z = {};
            acc[i][j] = z;
        }

    // depthwise work assignment: one channel-in-chunk, 8 consecutive pixels
    const int cOff  = tid >> 3;        // 0..31
    const int nBase = (tid & 7) * 8;   // 0,8,...,56

    for (int kc = 0; kc < CIN; kc += KC) {
        // ---- stage A chunk: Alds[m][k] = f16(pw[m][kc+k] + delta[b][m][kc+k])
        {
            const int m = tid;
            const float* pwRow = pw    + (size_t)m * CIN + kc;
            const float* dRow  = delta + ((size_t)b * COUT + m) * CIN + kc;
            _Float16* arow = Alds + m * LDA;
            #pragma unroll
            for (int k = 0; k < KC; ++k)
                arow[k] = (_Float16)(pwRow[k] + dRow[k]);
            if (kc + KC < CIN) {
                __builtin_prefetch(pwRow + KC, 0, 1);
                __builtin_prefetch(dRow + KC, 0, 1);
            }
        }
        // ---- stage B chunk: depthwise 3x3 for channel kc+cOff, pixels nBase..nBase+7
        {
            const int c = kc + cOff;
            float k9[9];
            #pragma unroll
            for (int i = 0; i < 9; ++i) k9[i] = dwk[c * 9 + i];
            const float* xc = x + (size_t)(b * CIN + c) * (HH * WW);
            float a8[8];
            #pragma unroll
            for (int p = 0; p < 8; ++p) a8[p] = 0.f;
            #pragma unroll
            for (int dy = 0; dy < 3; ++dy) {
                const int y = h + dy - 1;
                if (y < 0 || y >= HH) continue;     // block-uniform branch
                const float* xrow = xc + (size_t)y * WW;
                float xv[10];
                #pragma unroll
                for (int i = 0; i < 10; ++i) {
                    const int xcol = w0 + nBase - 1 + i;
                    xv[i] = (xcol >= 0 && xcol < WW) ? xrow[xcol] : 0.f;
                }
                #pragma unroll
                for (int dx = 0; dx < 3; ++dx) {
                    const float kv = k9[dy * 3 + dx];
                    #pragma unroll
                    for (int p = 0; p < 8; ++p) a8[p] += kv * xv[p + dx];
                }
            }
            #pragma unroll
            for (int p = 0; p < 8; ++p)
                Blds[(nBase + p) * LDB + cOff] = (_Float16)a8[p];
        }
        __syncthreads();

        // ---- B fragments (32x16 f16): lane = column n; K {0-7,16-23} / {8-15,24-31} by lane half
        v16h bfrag[2];
        #pragma unroll
        for (int nj = 0; nj < 2; ++nj) {
            const int n = nWave + nj * 16 + laneLo;
            const _Float16* bp = Blds + n * LDB + laneHi * 8;
            half8 lo = *(const half8*)(bp);
            half8 hi = *(const half8*)(bp + 16);
            v16h bf;
            #pragma unroll
            for (int i = 0; i < 8; ++i) { bf[i] = lo[i]; bf[8 + i] = hi[i]; }
            bfrag[nj] = bf;
        }
        // ---- A fragments (16x32 f16): lane = row m; K {0-7,16-23} / {8-15,24-31} by lane half
        #pragma unroll
        for (int mi = 0; mi < 4; ++mi) {
            const int m = mWave + mi * 16 + laneLo;
            const _Float16* arow = Alds + m * LDA;
            half8 lo = *(const half8*)(arow + laneHi * 8);
            half8 hi = *(const half8*)(arow + 16 + laneHi * 8);
            v16h af;
            #pragma unroll
            for (int i = 0; i < 8; ++i) { af[i] = lo[i]; af[8 + i] = hi[i]; }
            #pragma unroll
            for (int nj = 0; nj < 2; ++nj) {
                acc[mi][nj] = __builtin_amdgcn_wmma_f32_16x16x32_f16(
                    false, af, false, bfrag[nj], (short)0, acc[mi][nj], false, false);
            }
        }
        __syncthreads();
    }

    // ---- store: C/D 16x16 f32 layout -> lane = N col, VGPR v = M row (+8 for hi half)
    #pragma unroll
    for (int mi = 0; mi < 4; ++mi) {
        #pragma unroll
        for (int nj = 0; nj < 2; ++nj) {
            const int nCol  = w0 + nWave + nj * 16 + laneLo;
            const int mRow0 = mWave + mi * 16 + laneHi * 8;
            float* op = out + (((size_t)b * COUT + mRow0) * HH + h) * WW + nCol;
            #pragma unroll
            for (int v = 0; v < 8; ++v)
                op[(size_t)v * HH * WW] = acc[mi][nj][v];
        }
    }
}

extern "C" void kernel_launch(void* const* d_in, const int* in_sizes, int n_in,
                              void* d_out, int out_size, void* d_ws, size_t ws_size,
                              hipStream_t stream) {
    const float* x     = (const float*)d_in[0]; // (8,256,128,128)
    const float* delta = (const float*)d_in[1]; // (8,256,256,1,1)
    const float* dwk   = (const float*)d_in[2]; // (256,1,3,3)
    const float* pw    = (const float*)d_in[3]; // (256,256,1,1)
    float* out = (float*)d_out;

    dim3 grid(HH * (WW / TILE_N), 8);  // (256, 8): one row-half tile per block
    dim3 block(256);
    fused_dw_pw_wmma<<<grid, block, 0, stream>>>(x, delta, dwk, pw, out);
}
